// MegatronMoEWrapper_44925357916272
// MI455X (gfx1250) — compile-verified, matrix operand order checked by reading, not run on hardware
//
#include <hip/hip_runtime.h>
#include <hip/hip_bf16.h>

#define N_TOKENS 2048
#define D_MODEL  1024
#define D_FF     4096
#define N_EXP    8

typedef __attribute__((ext_vector_type(16))) __bf16 v16bf;
typedef __attribute__((ext_vector_type(8)))  __bf16 v8bf;
typedef __attribute__((ext_vector_type(8)))  float  v8f;

// ---------------------------------------------------------------------------
// Fragment builders following the CDNA5 16-bit WMMA operand layout:
// lanes 0-15 hold K 0..7 and 16..23, lanes 16-31 hold K 8..15 and 24..31.
// ---------------------------------------------------------------------------
__device__ __forceinline__ v16bf frag_bf16(const __bf16* p, int lane) {
  const int base = (lane & 16) ? 8 : 0;
  v8bf lo = *(const v8bf*)(p + base);
  v8bf hi = *(const v8bf*)(p + base + 16);
  return __builtin_shufflevector(lo, hi, 0,1,2,3,4,5,6,7,8,9,10,11,12,13,14,15);
}

__device__ __forceinline__ v16bf frag_from_f32(const float* __restrict__ p, int lane) {
  const int base = (lane & 16) ? 8 : 0;
  float4 a0 = *(const float4*)(p + base);
  float4 a1 = *(const float4*)(p + base + 4);
  float4 a2 = *(const float4*)(p + base + 16);
  float4 a3 = *(const float4*)(p + base + 20);
  v16bf b;
  b[0]=(__bf16)a0.x;  b[1]=(__bf16)a0.y;  b[2]=(__bf16)a0.z;  b[3]=(__bf16)a0.w;
  b[4]=(__bf16)a1.x;  b[5]=(__bf16)a1.y;  b[6]=(__bf16)a1.z;  b[7]=(__bf16)a1.w;
  b[8]=(__bf16)a2.x;  b[9]=(__bf16)a2.y;  b[10]=(__bf16)a2.z; b[11]=(__bf16)a2.w;
  b[12]=(__bf16)a3.x; b[13]=(__bf16)a3.y; b[14]=(__bf16)a3.z; b[15]=(__bf16)a3.w;
  return b;
}

// ---------------------------------------------------------------------------
// K1: gating. gate_w staged in LDS (8x1024 f32 = 32KB), one thread per token.
// Top-2 + 2-way softmax; atomic append into per-expert entry lists.
// entry = token*2 + slot; gwbuf[entry] = gate weight.
// ---------------------------------------------------------------------------
__global__ void moe_gate_kernel(const float* __restrict__ x,
                                const float* __restrict__ gate_w,
                                int* __restrict__ counts,
                                int* __restrict__ entries,
                                float* __restrict__ gwbuf) {
  __shared__ float gl[N_EXP * D_MODEL];
  const int tid = threadIdx.x;
  for (int i = tid; i < N_EXP * D_MODEL; i += 256) gl[i] = gate_w[i];
  __syncthreads();

  const int t = blockIdx.x * 256 + tid;
  const float* xr = x + (size_t)t * D_MODEL;
  float acc[N_EXP];
#pragma unroll
  for (int e = 0; e < N_EXP; ++e) acc[e] = 0.f;

  for (int d = 0; d < D_MODEL; d += 4) {
    float4 xv = *(const float4*)(xr + d);
#pragma unroll
    for (int e = 0; e < N_EXP; ++e) {
      const float* g = gl + e * D_MODEL + d;
      acc[e] += xv.x * g[0] + xv.y * g[1] + xv.z * g[2] + xv.w * g[3];
    }
  }

  int e0 = 0; float v0 = acc[0];
#pragma unroll
  for (int e = 1; e < N_EXP; ++e) if (acc[e] > v0) { v0 = acc[e]; e0 = e; }
  int e1 = -1; float v1 = -3.4e38f;
#pragma unroll
  for (int e = 0; e < N_EXP; ++e) if (e != e0 && acc[e] > v1) { v1 = acc[e]; e1 = e; }

  const float p0 = 1.f / (1.f + expf(v1 - v0));   // softmax over the two logits
  const float p1 = 1.f - p0;

  int pos0 = atomicAdd(&counts[e0], 1);
  entries[e0 * N_TOKENS + pos0] = t * 2;
  gwbuf[t * 2] = p0;
  int pos1 = atomicAdd(&counts[e1], 1);
  entries[e1 * N_TOKENS + pos1] = t * 2 + 1;
  gwbuf[t * 2 + 1] = p1;
}

// ---------------------------------------------------------------------------
// Streaming f32 -> bf16 conversion (used for x, w1, w2). 8 elems/thread.
// ---------------------------------------------------------------------------
__global__ void cvt_bf16_kernel(const float* __restrict__ in,
                                __bf16* __restrict__ out) {
  const size_t i = ((size_t)blockIdx.x * 256 + threadIdx.x) * 8;
  float4 a = *(const float4*)(in + i);
  float4 b = *(const float4*)(in + i + 4);
  v8bf o;
  o[0]=(__bf16)a.x; o[1]=(__bf16)a.y; o[2]=(__bf16)a.z; o[3]=(__bf16)a.w;
  o[4]=(__bf16)b.x; o[5]=(__bf16)b.y; o[6]=(__bf16)b.z; o[7]=(__bf16)b.w;
  *(v8bf*)(out + i) = o;
}

// ---------------------------------------------------------------------------
// K2: fused expert FFN on WMMA. Block = 256 threads (8 waves), one block per
// (expert, 16-token tile). Per 128-wide d_ff chunk:
//   stage 1: wave w computes h[16 x 16] = x_tile @ w1[f0+16w : +16]^T (K=1024)
//            -> exact-erf GELU -> bf16 h tile in LDS
//   stage 2: wave w accumulates y[16 x 128] slice with A = h (K=128)
// BF16W: weights pre-converted to bf16 (no cvt in the hot loop);
// else f32 weights converted in-register (fallback when ws is small).
// ---------------------------------------------------------------------------
template<bool BF16W>
__global__ void __launch_bounds__(256, 1)
moe_expert_kernel(const float* __restrict__ w1,
                  const float* __restrict__ w2,
                  const __bf16* __restrict__ wb1,
                  const __bf16* __restrict__ wb2,
                  const __bf16* __restrict__ xb,
                  const int* __restrict__ entries,
                  const float* __restrict__ gwbuf,
                  const int* __restrict__ counts,
                  float* __restrict__ slots) {
  __shared__ __bf16 lds_x[16 * D_MODEL];  // 32 KB
  __shared__ __bf16 lds_h[16 * 128];      // 4 KB
  __shared__ int    lds_ent[16];
  __shared__ float  lds_gw[16];

  const int e    = blockIdx.x >> 7;     // 128 blocks per expert
  const int blk  = blockIdx.x & 127;
  const int rows = counts[e];
  const int row0 = blk * 16;
  if (row0 >= rows) return;

  const int tid  = threadIdx.x;
  const int wave = tid >> 5;
  const int lane = tid & 31;
  const int lm   = lane & 15;
  const int half = (lane & 16) ? 8 : 0;

  const float*  w1e  = w1  + (size_t)e * D_FF * D_MODEL;
  const float*  w2e  = w2  + (size_t)e * D_MODEL * D_FF;
  const __bf16* wb1e = wb1 + (size_t)e * D_FF * D_MODEL;
  const __bf16* wb2e = wb2 + (size_t)e * D_MODEL * D_FF;

  if (tid < 16) {
    int rg = row0 + tid;
    int ent = 0; float gv = 0.f;
    if (rg < rows) { ent = entries[e * N_TOKENS + rg]; gv = gwbuf[ent]; }
    lds_ent[tid] = ent;
    lds_gw[tid]  = gv;
  }
  __syncthreads();

  // Gather 16 token rows of bf16 x into LDS (each thread copies 128B).
  {
    const int r = tid >> 4, c = tid & 15;
    const uint4* src = (const uint4*)(xb + (size_t)(lds_ent[r] >> 1) * D_MODEL + c * 64);
    uint4* dst = (uint4*)(lds_x + r * D_MODEL + c * 64);
#pragma unroll
    for (int i = 0; i < 8; ++i) dst[i] = src[i];
  }
  __syncthreads();

  v8f acc[8];
#pragma unroll
  for (int n = 0; n < 8; ++n)
#pragma unroll
    for (int r = 0; r < 8; ++r) acc[n][r] = 0.f;

  for (int f0 = 0; f0 < D_FF; f0 += 128) {
    // ---- stage 1: h tile (one 16x16 per wave), K = D_MODEL ----
    v8f c1;
#pragma unroll
    for (int r = 0; r < 8; ++r) c1[r] = 0.f;
    const int frow = f0 + wave * 16 + lm;
    const float*  w1row  = w1e  + (size_t)frow * D_MODEL;
    const __bf16* wb1row = wb1e + (size_t)frow * D_MODEL;
#pragma unroll 4
    for (int kk = 0; kk < D_MODEL; kk += 32) {
      v16bf a = frag_bf16(lds_x + lm * D_MODEL + kk, lane);
      v16bf b = BF16W ? frag_bf16(wb1row + kk, lane)
                      : frag_from_f32(w1row + kk, lane);
      c1 = __builtin_amdgcn_wmma_f32_16x16x32_bf16(false, a, false, b,
                                                   (short)0, c1, false, false);
    }

    __syncthreads();  // previous chunk's stage-2 reads of lds_h complete
#pragma unroll
    for (int r = 0; r < 8; ++r) {
      float v = c1[r];
      float g = 0.5f * v * (1.f + erff(v * 0.70710678118654752f));  // exact GELU
      lds_h[(r + half) * 128 + wave * 16 + lm] = (__bf16)g;
    }
    __syncthreads();  // h tile visible

    // ---- stage 2: y slice accumulate, K = 128 (this chunk) ----
#pragma unroll
    for (int kk = 0; kk < 128; kk += 32) {
      v16bf a = frag_bf16(lds_h + lm * 128 + kk, lane);
#pragma unroll
      for (int nt = 0; nt < 8; ++nt) {
        const int col = wave * 128 + nt * 16 + lm;
        v16bf b = BF16W ? frag_bf16(wb2e + (size_t)col * D_FF + f0 + kk, lane)
                        : frag_from_f32(w2e + (size_t)col * D_FF + f0 + kk, lane);
        acc[nt] = __builtin_amdgcn_wmma_f32_16x16x32_bf16(false, a, false, b,
                                                          (short)0, acc[nt],
                                                          false, false);
      }
    }
  }

  // Scale by gate weight, scatter to per-(token,slot) buffer.
#pragma unroll
  for (int nt = 0; nt < 8; ++nt) {
#pragma unroll
    for (int r = 0; r < 8; ++r) {
      const int M = r + half;
      if (row0 + M < rows) {
        const int col = wave * 128 + nt * 16 + lm;
        slots[(size_t)lds_ent[M] * D_MODEL + col] = acc[nt][r] * lds_gw[M];
      }
    }
  }
}

// ---------------------------------------------------------------------------
// K3: out[t] = slots[2t] + slots[2t+1], float4-vectorized.
// ---------------------------------------------------------------------------
__global__ void moe_combine_kernel(const float* __restrict__ slots,
                                   float* __restrict__ out) {
  const int i4 = blockIdx.x * 256 + threadIdx.x;     // float4 index
  const int d4 = i4 & 255;                           // 1024/4 per row
  const int t  = i4 >> 8;
  const float4* s = (const float4*)slots;
  float4 a = s[(size_t)(2 * t) * 256 + d4];
  float4 b = s[(size_t)(2 * t + 1) * 256 + d4];
  float4 o; o.x = a.x + b.x; o.y = a.y + b.y; o.z = a.z + b.z; o.w = a.w + b.w;
  ((float4*)out)[i4] = o;
}

// ---------------------------------------------------------------------------
// Host launcher. ws layout (bytes):
//   [0,32)          counts[8]          (memset to 0 each launch)
//   [256,65792)     entries[8][2048]
//   [65792,82176)   gwbuf[4096]
//   [82176,+4MB)    xb (bf16 x)
//   [4276480,+16MB) slots[4096][1024] f32
//   [21053696,+64MB) wb1 (bf16 w1)    } only when ws_size permits
//   [88162560,+64MB) wb2 (bf16 w2)    }
// ---------------------------------------------------------------------------
extern "C" void kernel_launch(void* const* d_in, const int* in_sizes, int n_in,
                              void* d_out, int out_size, void* d_ws, size_t ws_size,
                              hipStream_t stream) {
  const float* x      = (const float*)d_in[0];
  const float* gate_w = (const float*)d_in[1];
  const float* w1     = (const float*)d_in[2];
  const float* w2     = (const float*)d_in[3];
  float* out = (float*)d_out;

  char* wsb       = (char*)d_ws;
  int*    counts  = (int*)wsb;
  int*    entries = (int*)(wsb + 256);
  float*  gwbuf   = (float*)(wsb + 65792);
  __bf16* xbf     = (__bf16*)(wsb + 82176);
  float*  slots   = (float*)(wsb + 4276480);
  __bf16* wb1     = (__bf16*)(wsb + 21053696);
  __bf16* wb2     = (__bf16*)(wsb + 88162560);

  const size_t need_bf16w = 155271424;          // end of wb2 region
  const bool use_bf16w = (ws_size >= need_bf16w);   // deterministic per run

  hipMemsetAsync(counts, 0, N_EXP * sizeof(int), stream);

  moe_gate_kernel<<<N_TOKENS / 256, 256, 0, stream>>>(x, gate_w, counts, entries, gwbuf);
  cvt_bf16_kernel<<<(N_TOKENS * D_MODEL) / (256 * 8), 256, 0, stream>>>(x, xbf);

  if (use_bf16w) {
    const int wblocks = (N_EXP * D_FF * D_MODEL) / (256 * 8);  // 16384
    cvt_bf16_kernel<<<wblocks, 256, 0, stream>>>(w1, wb1);
    cvt_bf16_kernel<<<wblocks, 256, 0, stream>>>(w2, wb2);
    moe_expert_kernel<true><<<N_EXP * 128, 256, 0, stream>>>(
        w1, w2, wb1, wb2, xbf, entries, gwbuf, counts, slots);
  } else {
    moe_expert_kernel<false><<<N_EXP * 128, 256, 0, stream>>>(
        w1, w2, wb1, wb2, xbf, entries, gwbuf, counts, slots);
  }

  moe_combine_kernel<<<(N_TOKENS * D_MODEL / 4) / 256, 256, 0, stream>>>(slots, out);
}
